// Attention_24292335027075
// MI455X (gfx1250) — compile-verified, matrix operand order checked by reading, not run on hardware
//
#include <hip/hip_runtime.h>
#include <hip/hip_bf16.h>

typedef __bf16 bf16_t;
typedef __attribute__((ext_vector_type(16))) __bf16 v16bf;
typedef __attribute__((ext_vector_type(8)))  float  v8f;
typedef __attribute__((ext_vector_type(4)))  int    v4i;

#define WMMA_BF16(A, B, C) \
  __builtin_amdgcn_wmma_f32_16x16x32_bf16(false, (A), false, (B), (short)0, (C), false, false)

#define AS1 __attribute__((address_space(1)))
#define AS3 __attribute__((address_space(3)))

#if defined(__has_builtin)
#  if __has_builtin(__builtin_amdgcn_global_load_async_to_lds_b128)
#    define HAVE_ASYNC_LDS 1
#  endif
#endif

// 16-byte global -> LDS copy; async (ASYNCcnt-tracked) when available.
__device__ __forceinline__ void copy16_to_lds(void* lds, const void* g) {
#if defined(HAVE_ASYNC_LDS)
    __builtin_amdgcn_global_load_async_to_lds_b128(
        (AS1 v4i*)(void*)g, (AS3 v4i*)(void*)lds, 0, 0);
#else
    *(uint4*)lds = *(const uint4*)g;
#endif
}
__device__ __forceinline__ void wait_lds_copies() {
#if defined(HAVE_ASYNC_LDS)
#  if __has_builtin(__builtin_amdgcn_s_wait_asynccnt)
    __builtin_amdgcn_s_wait_asynccnt(0);
#  else
    asm volatile("s_wait_asynccnt 0" ::: "memory");
#  endif
#endif
}

static constexpr int Bsz = 4;
static constexpr int Seq = 1024;
static constexpr int Dim = 1024;
static constexpr int NH  = 16;
static constexpr int HDm = 64;

// ---------------------------------------------------------------------------
// fp32 -> bf16 elementwise (x)
// ---------------------------------------------------------------------------
__global__ void cvt_bf16_kernel(const float* __restrict__ in, bf16_t* __restrict__ out, int n) {
    int i = blockIdx.x * blockDim.x + threadIdx.x;
    int stride = gridDim.x * blockDim.x;
    for (; i < n; i += stride) out[i] = (bf16_t)in[i];
}

// ---------------------------------------------------------------------------
// fp32 [R][C] -> bf16 [C][R] tiled transpose (weights), 32x32 tiles, 256 thr
// ---------------------------------------------------------------------------
__global__ void __launch_bounds__(256)
cvt_transpose_kernel(const float* __restrict__ in, bf16_t* __restrict__ out, int R, int C) {
    __shared__ bf16_t tile[32][33];
    const int c0 = blockIdx.x * 32, r0 = blockIdx.y * 32;
    const int tx = threadIdx.x & 31, ty = threadIdx.x >> 5;   // 32 x 8
    #pragma unroll
    for (int i = 0; i < 32; i += 8)
        tile[ty + i][tx] = (bf16_t)in[(size_t)(r0 + ty + i) * C + c0 + tx];
    __syncthreads();
    #pragma unroll
    for (int i = 0; i < 32; i += 8)
        out[(size_t)(c0 + ty + i) * R + r0 + tx] = tile[tx][ty + i];
}

// ---------------------------------------------------------------------------
// QKV GEMM: [4096,1024] x W^T[3072,1024] + b_attn
// 64(M)x128(N) block tiles, 8 waves, wave tile 16x64, double-buffered LDS
// fed by async global->LDS b128 copies. Epilogue scatters q (pre-scaled),
// k, v-transposed (bf16) and k,v (f32 `present`).
// ---------------------------------------------------------------------------
__global__ void __launch_bounds__(256)
qkv_gemm_kernel(const bf16_t* __restrict__ xb, const bf16_t* __restrict__ wabT,
                const float* __restrict__ b_attn,
                bf16_t* __restrict__ qb, bf16_t* __restrict__ kb, bf16_t* __restrict__ vbT,
                float* __restrict__ pk, float* __restrict__ pv)
{
    __shared__ __align__(32) bf16_t ldsA[2][64][32];    // [m][k]
    __shared__ __align__(32) bf16_t ldsB[2][128][32];   // [n][k]

    const int t    = threadIdx.x;
    const int wid  = t >> 5;
    const int lane = t & 31;
    const int row0 = blockIdx.x * 64;
    const int col0 = blockIdx.y * 128;
    const int m0   = (wid & 3) * 16;
    const int n0   = (wid >> 2) * 64;
    const int lm   = lane & 15;
    const int lk   = (lane >> 4) * 16;
    const int mlo  = (lane >> 4) * 8;

    v8f acc[4] = {};

    auto load_tiles = [&](int buf, int k0) {
        {   // A: 64x32
            int r = t >> 2, c = (t & 3) * 8;
            copy16_to_lds(&ldsA[buf][r][c], xb + (size_t)(row0 + r) * Dim + k0 + c);
        }
        #pragma unroll
        for (int u = 0; u < 2; ++u) {   // B: 128x32 (pre-transposed weights)
            int idx = t + u * 256;
            int n = idx >> 2, c = (idx & 3) * 8;
            copy16_to_lds(&ldsB[buf][n][c], wabT + (size_t)(col0 + n) * Dim + k0 + c);
        }
    };
    auto compute = [&](int buf) {
        v16bf afrag = *(const v16bf*)&ldsA[buf][m0 + lm][lk];
        #pragma unroll
        for (int f = 0; f < 4; ++f) {
            v16bf bfrag = *(const v16bf*)&ldsB[buf][n0 + f * 16 + lm][lk];
            acc[f] = WMMA_BF16(afrag, bfrag, acc[f]);
        }
    };

    load_tiles(0, 0);
    wait_lds_copies();
    __syncthreads();
    for (int k0 = 0; k0 < Dim; k0 += 64) {
        load_tiles(1, k0 + 32);
        compute(0);
        wait_lds_copies();
        __syncthreads();
        if (k0 + 64 < Dim) load_tiles(0, k0 + 64);
        compute(1);
        wait_lds_copies();
        __syncthreads();
    }

    #pragma unroll
    for (int f = 0; f < 4; ++f) {
        const int nc0 = col0 + n0 + f * 16;     // uniform per fragment
        const int sec = nc0 >> 10;              // 0=q 1=k 2=v (uniform branch)
        const int d0  = nc0 & 1023;
        const int h   = d0 >> 6;
        const int di  = (d0 & 63) + lm;
        const float bias = b_attn[nc0 + lm];
        #pragma unroll
        for (int j = 0; j < 8; ++j) {
            const int m  = row0 + m0 + mlo + j;
            const int bb = m >> 10;
            const int si = m & 1023;
            const int bh = bb * NH + h;
            const float v = acc[f][j] + bias;
            if (sec == 0) {
                qb[((size_t)bh * Seq + si) * HDm + di] = (bf16_t)(v * 0.125f);
            } else if (sec == 1) {
                const size_t hidx = ((size_t)bh * Seq + si) * HDm + di;
                kb[hidx] = (bf16_t)v;
                pk[hidx] = v;
            } else {
                vbT[((size_t)bh * HDm + di) * Seq + si] = (bf16_t)v;   // head-transposed
                pv[((size_t)bh * Seq + si) * HDm + di]  = v;
            }
        }
    }
}

// ---------------------------------------------------------------------------
// Flash attention. block = (b*H, 64 q-rows), 128 thr = 4 waves x 16 rows.
// K and transposed-V fragments stream straight from global (contiguous 32B
// per lane); probs round-trip through wave-private LDS. No block barriers.
// ---------------------------------------------------------------------------
__global__ void __launch_bounds__(128)
attn_kernel(const bf16_t* __restrict__ qb, const bf16_t* __restrict__ kb,
            const bf16_t* __restrict__ vbT, const float* __restrict__ cls_mask,
            bf16_t* __restrict__ ab)
{
    __shared__ __align__(32) bf16_t ldsP[4][16][32];   // per-wave probs [m][k]

    const int t    = threadIdx.x;
    const int wid  = t >> 5;
    const int lane = t & 31;
    const int bh   = blockIdx.x;
    const int b    = bh >> 4;
    const int h    = bh & 15;
    const int q0   = blockIdx.y * 64 + wid * 16;
    const int lm   = lane & 15;
    const int lk   = (lane >> 4) * 16;
    const int mlo  = (lane >> 4) * 8;

    const bf16_t* qbase = qb  + (size_t)bh * Seq * HDm;
    const bf16_t* kbase = kb  + (size_t)bh * Seq * HDm;
    const bf16_t* vtb   = vbT + (size_t)bh * HDm * Seq;
    const float*  cls0  = cls_mask + (size_t)b * 2 * Seq;
    const float*  cls1  = cls0 + Seq;

    const v16bf qa0 = *(const v16bf*)(qbase + (size_t)(q0 + lm) * HDm + lk);
    const v16bf qa1 = *(const v16bf*)(qbase + (size_t)(q0 + lm) * HDm + 32 + lk);

    v8f acc[4] = {};
    float mrun[8], lrun[8], alpha[8];
    #pragma unroll
    for (int j = 0; j < 8; ++j) { mrun[j] = -1e30f; lrun[j] = 0.f; }

    for (int kt = 0; kt < Seq; kt += 32) {
        if (kt + 32 < Seq)   // speculative prefetch of next K tile
            __builtin_prefetch(kbase + (size_t)(kt + 32 + lm) * HDm + lk, 0, 0);

        // scores S[16q x 32k] = Q(16x64) * K^T(64x32)
        v8f s0 = {}, s1 = {};
        {
            const bf16_t* kp0 = kbase + (size_t)(kt + lm) * HDm + lk;
            v16bf kf0 = *(const v16bf*)(kp0);
            v16bf kf1 = *(const v16bf*)(kp0 + 32);
            s0 = WMMA_BF16(qa0, kf0, s0);
            s0 = WMMA_BF16(qa1, kf1, s0);
            const bf16_t* kp1 = kbase + (size_t)(kt + 16 + lm) * HDm + lk;
            v16bf kg0 = *(const v16bf*)(kp1);
            v16bf kg1 = *(const v16bf*)(kp1 + 32);
            s1 = WMMA_BF16(qa0, kg0, s1);
            s1 = WMMA_BF16(qa1, kg1, s1);
        }

        const int kc0 = kt + lm;
        const int kc1 = kc0 + 16;
        #pragma unroll
        for (int j = 0; j < 8; ++j) {
            const int qrow = q0 + mlo + j;
            float v0 = s0[j], v1 = s1[j];
            float cm0 = (kc0 == 0) ? cls1[qrow]
                      : ((qrow == 0) ? cls0[kc0] : ((kc0 <= qrow) ? 1.f : 0.f));
            float cm1 = (kc1 == 0) ? cls1[qrow]
                      : ((qrow == 0) ? cls0[kc1] : ((kc1 <= qrow) ? 1.f : 0.f));
            v0 = v0 * cm0 - 10000.f * (1.f - cm0);
            v1 = v1 * cm1 - 10000.f * (1.f - cm1);

            float tmax = fmaxf(v0, v1);
            #pragma unroll
            for (int msk = 1; msk < 16; msk <<= 1)
                tmax = fmaxf(tmax, __shfl_xor(tmax, msk, 32));
            const float mnew = fmaxf(mrun[j], tmax);
            const float al   = __expf(mrun[j] - mnew);
            const float p0   = __expf(v0 - mnew);
            const float p1   = __expf(v1 - mnew);
            float ps = p0 + p1;
            #pragma unroll
            for (int msk = 1; msk < 16; msk <<= 1)
                ps += __shfl_xor(ps, msk, 32);
            lrun[j]  = lrun[j] * al + ps;
            mrun[j]  = mnew;
            alpha[j] = al;

            const int r = mlo + j;
            ldsP[wid][r][lm]      = (bf16_t)p0;
            ldsP[wid][r][16 + lm] = (bf16_t)p1;
        }

        #pragma unroll
        for (int f = 0; f < 4; ++f)
            #pragma unroll
            for (int j = 0; j < 8; ++j) acc[f][j] *= alpha[j];

        // O += P(16x32) * V(32x64); V fragments direct from transposed global V
        v16bf pa = *(const v16bf*)&ldsP[wid][lm][lk];
        #pragma unroll
        for (int f = 0; f < 4; ++f) {
            v16bf vf = *(const v16bf*)(vtb + (size_t)(f * 16 + lm) * Seq + kt + lk);
            acc[f] = WMMA_BF16(pa, vf, acc[f]);
        }
    }

    #pragma unroll
    for (int f = 0; f < 4; ++f) {
        #pragma unroll
        for (int j = 0; j < 8; ++j) {
            const int qrow = q0 + mlo + j;
            const float o  = acc[f][j] / lrun[j];
            ab[((size_t)(b * Seq + qrow)) * Dim + h * HDm + f * 16 + lm] = (bf16_t)o;
        }
    }
}

// ---------------------------------------------------------------------------
// Proj GEMM: [4096,1024] x Wp^T[1024,1024] + b_proj -> d_out (f32)
// ---------------------------------------------------------------------------
__global__ void __launch_bounds__(256)
proj_gemm_kernel(const bf16_t* __restrict__ ab, const bf16_t* __restrict__ wpbT,
                 const float* __restrict__ b_proj, float* __restrict__ out)
{
    __shared__ __align__(32) bf16_t ldsA[2][64][32];
    __shared__ __align__(32) bf16_t ldsB[2][128][32];

    const int t    = threadIdx.x;
    const int wid  = t >> 5;
    const int lane = t & 31;
    const int row0 = blockIdx.x * 64;
    const int col0 = blockIdx.y * 128;
    const int m0   = (wid & 3) * 16;
    const int n0   = (wid >> 2) * 64;
    const int lm   = lane & 15;
    const int lk   = (lane >> 4) * 16;
    const int mlo  = (lane >> 4) * 8;

    v8f acc[4] = {};

    auto load_tiles = [&](int buf, int k0) {
        {
            int r = t >> 2, c = (t & 3) * 8;
            copy16_to_lds(&ldsA[buf][r][c], ab + (size_t)(row0 + r) * Dim + k0 + c);
        }
        #pragma unroll
        for (int u = 0; u < 2; ++u) {
            int idx = t + u * 256;
            int n = idx >> 2, c = (idx & 3) * 8;
            copy16_to_lds(&ldsB[buf][n][c], wpbT + (size_t)(col0 + n) * Dim + k0 + c);
        }
    };
    auto compute = [&](int buf) {
        v16bf afrag = *(const v16bf*)&ldsA[buf][m0 + lm][lk];
        #pragma unroll
        for (int f = 0; f < 4; ++f) {
            v16bf bfrag = *(const v16bf*)&ldsB[buf][n0 + f * 16 + lm][lk];
            acc[f] = WMMA_BF16(afrag, bfrag, acc[f]);
        }
    };

    load_tiles(0, 0);
    wait_lds_copies();
    __syncthreads();
    for (int k0 = 0; k0 < Dim; k0 += 64) {
        load_tiles(1, k0 + 32);
        compute(0);
        wait_lds_copies();
        __syncthreads();
        if (k0 + 64 < Dim) load_tiles(0, k0 + 64);
        compute(1);
        wait_lds_copies();
        __syncthreads();
    }

    #pragma unroll
    for (int f = 0; f < 4; ++f) {
        const int nc = col0 + n0 + f * 16 + lm;
        const float bias = b_proj[nc];
        #pragma unroll
        for (int j = 0; j < 8; ++j) {
            const int m = row0 + m0 + mlo + j;
            out[(size_t)m * Dim + nc] = acc[f][j] + bias;
        }
    }
}

// ---------------------------------------------------------------------------
extern "C" void kernel_launch(void* const* d_in, const int* in_sizes, int n_in,
                              void* d_out, int out_size, void* d_ws, size_t ws_size,
                              hipStream_t stream) {
    const float* x      = (const float*)d_in[0];
    const float* cls    = (const float*)d_in[1];
    const float* w_attn = (const float*)d_in[2];
    const float* b_attn = (const float*)d_in[3];
    const float* w_proj = (const float*)d_in[4];
    const float* b_proj = (const float*)d_in[5];

    float* out = (float*)d_out;
    float* pk  = out + (size_t)Bsz * Seq * Dim;
    float* pv  = pk  + (size_t)Bsz * NH * Seq * HDm;

    const size_t nX  = (size_t)Bsz * Seq * Dim;
    const size_t nWA = (size_t)Dim * 3 * Dim;
    const size_t nWP = (size_t)Dim * Dim;
    const size_t nHV = (size_t)Bsz * NH * Seq * HDm;

    bf16_t* xb   = (bf16_t*)d_ws;
    bf16_t* wabT = xb   + nX;     // [3072][1024]
    bf16_t* wpbT = wabT + nWA;    // [1024][1024]
    bf16_t* qb   = wpbT + nWP;    // [B,H,S,hd]
    bf16_t* kb   = qb   + nHV;    // [B,H,S,hd]
    bf16_t* vbT  = kb   + nHV;    // [B,H,hd,S]
    bf16_t* ab   = vbT  + nHV;    // [B*S][D]

    cvt_bf16_kernel<<<2048, 256, 0, stream>>>(x, xb, (int)nX);
    cvt_transpose_kernel<<<dim3(3 * Dim / 32, Dim / 32), 256, 0, stream>>>(w_attn, wabT, Dim, 3 * Dim);
    cvt_transpose_kernel<<<dim3(Dim / 32, Dim / 32),     256, 0, stream>>>(w_proj, wpbT, Dim, Dim);

    qkv_gemm_kernel<<<dim3(64, 24), 256, 0, stream>>>(xb, wabT, b_attn, qb, kb, vbT, pk, pv);
    attn_kernel<<<dim3(Bsz * NH, Seq / 64), 128, 0, stream>>>(qb, kb, vbT, cls, ab);
    proj_gemm_kernel<<<dim3(64, 8), 256, 0, stream>>>(ab, wpbT, b_proj, out);
}
